// EGNN_network_72000831750800
// MI455X (gfx1250) — matrix operand order repaired and checked by reading, not compiled, not run on hardware
//
#include <hip/hip_runtime.h>

#define BATCH 2
#define NN    1024
#define NW    8            // waves per block
#define TPB   (NW * 32)

typedef __attribute__((ext_vector_type(16))) _Float16 v16h;
typedef __attribute__((ext_vector_type(8)))  _Float16 v8h;
typedef __attribute__((ext_vector_type(8)))  float    v8f;
typedef __attribute__((ext_vector_type(8)))  int      v8i;
typedef __attribute__((ext_vector_type(4)))  int      v4i;

__device__ __forceinline__ float silu_f(float x) {
    // x * sigmoid(x): v_exp_f32 + v_rcp_f32 fast path
    return x * __builtin_amdgcn_rcpf(1.0f + __expf(-x));
}

// lane <-> lane^16 swap as a single VALU op (v_permlanex16_b32, identity selects)
__device__ __forceinline__ int swap16i(int x) {
    return __builtin_amdgcn_permlanex16(x, x, 0x76543210, 0xFEDCBA98, false, false);
}
__device__ __forceinline__ float swap16f(float x) {
    return __int_as_float(swap16i(__float_as_int(x)));
}

__global__ __launch_bounds__(TPB) void egnn_layer(
    const float* __restrict__ t,         // [B,N]
    const float* __restrict__ coors_in,  // [B,N,3]
    const float* __restrict__ vel_in,    // [B,N,3]
    const float* __restrict__ We1,       // [7,16]  (layer slice)
    const float* __restrict__ be1,       // [16]
    const float* __restrict__ We2,       // [16,16]
    const float* __restrict__ be2,       // [16]
    const float* __restrict__ Wc1,       // [16,16]
    const float* __restrict__ bc1,       // [16]
    const float* __restrict__ Wc2,       // [16]
    const float* __restrict__ bc2,       // [1]
    const float* __restrict__ Wv1,       // [3,16]
    const float* __restrict__ bv1,       // [16]
    const float* __restrict__ Wv2,       // [16]
    const float* __restrict__ bv2,       // [1]
    float* __restrict__ coors_out,       // [B,N,3]
    float* __restrict__ vel_out)         // [B,N,3]
{
    __shared__ float4 s_node[NN];        // (cx, cy, cz, t)
    __shared__ float  s_we2[256];
    __shared__ float  s_wc1[256];
    __shared__ float4 s_coef[16];        // {A1,B1,C1,be1} per feature
    __shared__ float  s_be2[16], s_bc1[16], s_wc2[16];
    __shared__ float  s_av[16], s_bv1[16], s_wv2[16];
    __shared__ float  s_acc[NW][32][3];  // both half-wave partials

    const int tid = threadIdx.x;
    const int b   = blockIdx.x / (NN / 16);
    const int i0  = (blockIdx.x % (NN / 16)) * 16;

    // ---- stage node data + fused weights into LDS --------------------------
    for (int n = tid; n < NN; n += TPB) {
        const int g = b * NN + n;
        s_node[n] = make_float4(coors_in[g * 3 + 0], coors_in[g * 3 + 1],
                                coors_in[g * 3 + 2], t[g]);
    }
    for (int k = tid; k < 256; k += TPB) { s_we2[k] = We2[k]; s_wc1[k] = Wc1[k]; }
    if (tid < 16) {
        const int f = tid;
        // e_in @ We1 with hi = [t_i,t_i,t_i], hj = [t_j,t_j,t_j], d2:
        s_coef[f] = make_float4(We1[f] + We1[16 + f] + We1[32 + f],        // coef t_i
                                We1[48 + f] + We1[64 + f] + We1[80 + f],   // coef t_j
                                We1[96 + f],                               // coef d2
                                be1[f]);
        s_be2[f] = be2[f]; s_bc1[f] = bc1[f]; s_wc2[f] = Wc2[f];
        s_av[f]  = Wv1[f] + Wv1[16 + f] + Wv1[32 + f];
        s_bv1[f] = bv1[f]; s_wv2[f] = Wv2[f];
    }
    __syncthreads();

    const int lane = tid & 31;
    const int wid  = tid >> 5;
    const int half = lane >> 4;          // lanes 0-15 own j0, lanes 16-31 own j1
    const int e    = lane & 15;          // i-node offset == B/D column index
    const int i    = i0 + e;
    const int kb   = half ? 8 : 0;       // A-layout K split per half-wave

    const float4 ni  = s_node[i];
    const float  ti  = ni.w;
    const float  cix = ni.x, ciy = ni.y, ciz = ni.z;

    // A matrices (Wᵀ) in 16-bit A-layout: lane m = out-feature,
    // contiguous K: lanes 0-15 hold K 0-15, lanes 16-31 hold K 16-31 (zero pad).
    v16h a_we2, a_wc1;
    #pragma unroll
    for (int k = 0; k < 8; ++k) {
        a_we2[k]     = (_Float16)(half ? 0.f : s_we2[k * 16 + e]);
        a_we2[k + 8] = (_Float16)(half ? 0.f : s_we2[(k + 8) * 16 + e]);
        a_wc1[k]     = (_Float16)(half ? 0.f : s_wc1[k * 16 + e]);
        a_wc1[k + 8] = (_Float16)(half ? 0.f : s_wc1[(k + 8) * 16 + e]);
    }
    v8f   c_e2, c_c1;
    float wc2h[8];
    #pragma unroll
    for (int r = 0; r < 8; ++r) {
        c_e2[r] = s_be2[r + 8 * half];
        c_c1[r] = s_bc1[r + 8 * half];
        wc2h[r] = s_wc2[r + 8 * half];
    }
    float4 cf[16];
    #pragma unroll
    for (int f = 0; f < 16; ++f) cf[f] = s_coef[f];
    const float bc2v = bc2[0];

    float ax = 0.f, ay = 0.f, az = 0.f;

    // ---- j loop: 2 j-nodes per iteration (one per half-wave), 32 edges ----
    for (int jp = wid; jp < NN / 2; jp += NW) {
        const int    jsel = 2 * jp + half;
        const float4 nj   = s_node[jsel];             // one ds_load_b128
        const float  tj   = nj.w;
        const float  rx   = cix - nj.x;
        const float  ry   = ciy - nj.y;
        const float  rz   = ciz - nj.z;
        const float  dd   = rx * rx + ry * ry + rz * rz;

        // m1 = silu(fused first layer) for this lane's own edge (i, jsel)
        v16h mh;
        #pragma unroll
        for (int f = 0; f < 16; ++f) {
            float x = fmaf(ti, cf[f].x, fmaf(tj, cf[f].y, fmaf(dd, cf[f].z, cf[f].w)));
            mh[f] = (_Float16)silu_f(x);
        }
        const v8i mp = __builtin_bit_cast(v8i, mh);

        // B matrices m1ᵀ[fi, i-col] for j0 (lower-half data) and j1 (swapped in)
        v8i b0i, b1i;
        #pragma unroll
        for (int r = 0; r < 8; ++r) {
            const int sw = swap16i(mp[r]);
            b0i[r] = half ? 0 : mp[r];
            b1i[r] = half ? 0 : sw;
        }

        const v8f d1a = __builtin_amdgcn_wmma_f32_16x16x32_f16(
            false, a_we2, false, __builtin_bit_cast(v16h, b0i), (short)0, c_e2, false, false);
        const v8f d1b = __builtin_amdgcn_wmma_f32_16x16x32_f16(
            false, a_we2, false, __builtin_bit_cast(v16h, b1i), (short)0, c_e2, false, false);

        // silu + f16 pack (lower lanes hold fi 0-7, upper lanes fi 8-15)
        v8h ha, hb;
        #pragma unroll
        for (int r = 0; r < 8; ++r) {
            ha[r] = (_Float16)silu_f(d1a[r]);
            hb[r] = (_Float16)silu_f(d1b[r]);
        }
        const v4i pa = __builtin_bit_cast(v4i, ha);
        const v4i pb = __builtin_bit_cast(v4i, hb);

        v8i b2a, b2b;
        #pragma unroll
        for (int r2 = 0; r2 < 4; ++r2) {
            const int swa = swap16i(pa[r2]);
            const int swb = swap16i(pb[r2]);
            b2a[r2]     = half ? 0 : pa[r2];
            b2a[r2 + 4] = half ? 0 : swa;
            b2b[r2]     = half ? 0 : pb[r2];
            b2b[r2 + 4] = half ? 0 : swb;
        }

        const v8f d2a = __builtin_amdgcn_wmma_f32_16x16x32_f16(
            false, a_wc1, false, __builtin_bit_cast(v16h, b2a), (short)0, c_c1, false, false);
        const v8f d2b = __builtin_amdgcn_wmma_f32_16x16x32_f16(
            false, a_wc1, false, __builtin_bit_cast(v16h, b2b), (short)0, c_c1, false, false);

        // w = silu(u) · Wc2 + bc2 : lane-local half-dot + one half-swap each
        float psa = 0.f, psb = 0.f;
        #pragma unroll
        for (int r = 0; r < 8; ++r) {
            psa = fmaf(silu_f(d2a[r]), wc2h[r], psa);
            psb = fmaf(silu_f(d2b[r]), wc2h[r], psb);
        }
        const float w0 = psa + swap16f(psa) + bc2v;
        const float w1 = psb + swap16f(psb) + bc2v;
        const float w  = half ? w1 : w0;     // each half accumulates its own j

        ax = fmaf(w, rx, ax);
        ay = fmaf(w, ry, ay);
        az = fmaf(w, rz, az);
    }

    s_acc[wid][lane][0] = ax;
    s_acc[wid][lane][1] = ay;
    s_acc[wid][lane][2] = az;
    __syncthreads();

    // ---- combine waves + halves, velocity gate, node update ----------------
    if (tid < 16) {
        const int ii = i0 + tid;
        float sx = 0.f, sy = 0.f, sz = 0.f;
        #pragma unroll
        for (int wv = 0; wv < NW; ++wv) {
            sx += s_acc[wv][tid][0] + s_acc[wv][tid + 16][0];
            sy += s_acc[wv][tid][1] + s_acc[wv][tid + 16][1];
            sz += s_acc[wv][tid][2] + s_acc[wv][tid + 16][2];
        }
        const float invN = 1.0f / (float)NN;
        sx *= invN; sy *= invN; sz *= invN;

        const float4 nii = s_node[ii];
        const float  tii = nii.w;
        float vw = bv2[0];
        #pragma unroll
        for (int f = 0; f < 16; ++f)
            vw = fmaf(silu_f(fmaf(tii, s_av[f], s_bv1[f])), s_wv2[f], vw);

        const int base = (b * NN + ii) * 3;
        const float nvx = fmaf(vw, vel_in[base + 0], sx);
        const float nvy = fmaf(vw, vel_in[base + 1], sy);
        const float nvz = fmaf(vw, vel_in[base + 2], sz);
        vel_out[base + 0] = nvx;
        vel_out[base + 1] = nvy;
        vel_out[base + 2] = nvz;
        coors_out[base + 0] = nii.x + nvx;
        coors_out[base + 1] = nii.y + nvy;
        coors_out[base + 2] = nii.z + nvz;
    }
}

extern "C" void kernel_launch(void* const* d_in, const int* in_sizes, int n_in,
                              void* d_out, int out_size, void* d_ws, size_t ws_size,
                              hipStream_t stream) {
    (void)in_sizes; (void)n_in; (void)out_size; (void)ws_size;
    const float* t     = (const float*)d_in[0];
    const float* coors = (const float*)d_in[1];
    const float* vel   = (const float*)d_in[2];
    const float* We1   = (const float*)d_in[3];
    const float* be1   = (const float*)d_in[4];
    const float* We2   = (const float*)d_in[5];
    const float* be2   = (const float*)d_in[6];
    const float* Wc1   = (const float*)d_in[7];
    const float* bc1   = (const float*)d_in[8];
    const float* Wc2   = (const float*)d_in[9];
    const float* bc2   = (const float*)d_in[10];
    const float* Wv1   = (const float*)d_in[11];
    const float* bv1   = (const float*)d_in[12];
    const float* Wv2   = (const float*)d_in[13];
    const float* bv2   = (const float*)d_in[14];

    const int nper = BATCH * NN * 3;      // 6144 floats
    float* ws = (float*)d_ws;
    float* cbuf[2] = { ws,        ws + 2 * nper };
    float* vbuf[2] = { ws + nper, ws + 3 * nper };
    float* co = (float*)d_out;
    float* vo = co + nper;                // output tuple: (coors, vel) concatenated

    dim3 grid(BATCH * (NN / 16));
    dim3 blk(TPB);

    for (int l = 0; l < 3; ++l) {
        const float* ci = (l == 0) ? coors : cbuf[(l - 1) & 1];
        const float* vi = (l == 0) ? vel   : vbuf[(l - 1) & 1];
        float* cout = (l == 2) ? co : cbuf[l & 1];
        float* vout = (l == 2) ? vo : vbuf[l & 1];
        egnn_layer<<<grid, blk, 0, stream>>>(
            t, ci, vi,
            We1 + l * 112, be1 + l * 16, We2 + l * 256, be2 + l * 16,
            Wc1 + l * 256, bc1 + l * 16, Wc2 + l * 16,  bc2 + l,
            Wv1 + l * 48,  bv1 + l * 16, Wv2 + l * 16,  bv2 + l,
            cout, vout);
    }
}
